// DrugGraphConv_53798760350030
// MI455X (gfx1250) — matrix operand order repaired and checked by reading, not compiled, not run on hardware
//
#include <hip/hip_runtime.h>
#include <hip/hip_bf16.h>

typedef __attribute__((ext_vector_type(16))) __bf16 v16bf;
typedef __attribute__((ext_vector_type(8)))  float  v8f;

#define NG   5000
#define NPG  100
#define EPG  400

// per-layer dims: d_in, d_out, K = 3*d_in padded to 32, N padded to 16
__device__ __constant__ const int kDIN[5]  = { 74,  70,  65,  60,  55};
__device__ __constant__ const int kDOUT[5] = { 70,  65,  60,  55,  37};
__device__ __constant__ const int kKPAD[5] = {224, 224, 224, 192, 192};
__device__ __constant__ const int kNPAD[5] = { 80,  80,  64,  64,  48};

#define FEAT_PITCH 224   // bf16 elems per row (448 B, 16B-aligned chunks)
#define ACC_PITCH  80

// LDS layout (bytes)
#define OFF_FEAT 0                               // 112*224 bf16 = 50176
#define OFF_WT   50176                           // 80*224  bf16 = 35840
#define OFF_ACC  86016                           // 112*80  f32  = 35840
#define OFF_BIAS 121856                          // 80 f32
#define OFF_NORM 122176                          // 112 f32
#define OFF_GATE 122624                          // 128 f32
#define OFF_IDEG 123136                          // 112 i32
#define OFF_ESRC 123584                          // 400 i16
#define OFF_EDST 124384                          // 400 i16
#define OFF_RED  125184                          // 4 f32
#define SMEM_BYTES 125248

union ABu { uint4 q[2]; v16bf v; };

__device__ __forceinline__ unsigned short f2bf(float f) {
    unsigned int u = __float_as_uint(f);
    u += 0x7FFFu + ((u >> 16) & 1u);          // round-to-nearest-even
    return (unsigned short)(u >> 16);
}
__device__ __forceinline__ float bf2f(unsigned short h) {
    return __uint_as_float(((unsigned int)h) << 16);
}

__global__ __launch_bounds__(256)
void DrugGraphConv_53798760350030_kernel(
    const float* __restrict__ x,
    const int*   __restrict__ src,
    const int*   __restrict__ dst,
    const float* __restrict__ W0, const float* __restrict__ b0,
    const float* __restrict__ W1, const float* __restrict__ b1,
    const float* __restrict__ W2, const float* __restrict__ b2,
    const float* __restrict__ W3, const float* __restrict__ b3,
    const float* __restrict__ W4, const float* __restrict__ b4,
    const float* __restrict__ gate_w, const float* __restrict__ gate_b,
    float* __restrict__ out)
{
    extern __shared__ unsigned char smem[];
    unsigned short* feat = (unsigned short*)(smem + OFF_FEAT);
    unsigned short* wT   = (unsigned short*)(smem + OFF_WT);
    float*          acc  = (float*)(smem + OFF_ACC);
    float*          bias = (float*)(smem + OFF_BIAS);
    float*          nrm  = (float*)(smem + OFF_NORM);
    float*          gate = (float*)(smem + OFF_GATE);
    int*            ideg = (int*)  (smem + OFF_IDEG);
    short*          esrc = (short*)(smem + OFF_ESRC);
    short*          edst = (short*)(smem + OFF_EDST);
    float*          red  = (float*)(smem + OFF_RED);

    const int g    = blockIdx.x;
    const int tid  = threadIdx.x;
    const int lane = tid & 31;
    const int wave = tid >> 5;
    const int hi   = lane >> 4;     // upper/lower half-wave
    const int lo   = lane & 15;

    const float* Wg[5] = {W0, W1, W2, W3, W4};
    const float* bg[5] = {b0, b1, b2, b3, b4};

    // ---- init: zero feature buffer, degrees ----
    for (int i = tid; i < 112 * FEAT_PITCH / 2; i += 256) ((unsigned int*)feat)[i] = 0u;
    for (int i = tid; i < 112; i += 256) ideg[i] = 0;
    __syncthreads();

    // ---- edges (local ids) + degree ----
    for (int e = tid; e < EPG; e += 256) {
        int s = src[g * EPG + e] - g * NPG;
        int d = dst[g * EPG + e] - g * NPG;
        esrc[e] = (short)s;
        edst[e] = (short)d;
        atomicAdd(&ideg[d], 1);
    }
    // ---- x -> bf16 feat block 0 ----
    for (int i = tid; i < NPG * 74; i += 256) {
        int m = i / 74, f = i - m * 74;
        feat[m * FEAT_PITCH + f] = f2bf(x[(g * NPG + m) * 74 + f]);
    }
    __syncthreads();
    for (int m = tid; m < 112; m += 256) {
        float dg = (m < NPG) ? (float)ideg[m] : 1.0f;
        nrm[m] = rsqrtf(fmaxf(dg, 1.0f));          // clip(deg,1)^-0.5
    }
    __syncthreads();

#pragma unroll
    for (int L = 0; L < 5; ++L) {
        const int din  = kDIN[L];
        const int dout = kDOUT[L];
        const int kpad = kKPAD[L];
        const int npad = kNPAD[L];

        // ---- K=2 propagation hops: h_{t+1} = norm * scatter_add(norm[src]*h_t[src]) ----
        for (int hop = 0; hop < 2; ++hop) {
            for (int i = tid; i < 112 * ACC_PITCH; i += 256) acc[i] = 0.f;
            __syncthreads();
            const int base = hop * din;
            for (int i = tid; i < EPG * din; i += 256) {
                int e = i / din, f = i - e * din;
                int s = esrc[e], d = edst[e];
                float v = bf2f(feat[s * FEAT_PITCH + base + f]) * nrm[s];
                atomicAdd(&acc[d * ACC_PITCH + f], v);       // ds_add_f32
            }
            __syncthreads();
            for (int i = tid; i < NPG * din; i += 256) {
                int m = i / din, f = i - m * din;
                feat[m * FEAT_PITCH + (hop + 1) * din + f] = f2bf(nrm[m] * acc[m * ACC_PITCH + f]);
            }
            __syncthreads();
        }

        // ---- stage weights: W[k][n] (f32, L2-resident) -> wT[n][k] bf16, zero-padded ----
        const float* W = Wg[L];
        const int k3 = 3 * din;
        for (int i = tid; i < npad * FEAT_PITCH; i += 256) wT[i] = 0;
        __syncthreads();
        for (int i = tid; i < k3 * dout; i += 256) {
            int k = i / dout, n = i - k * dout;
            wT[n * FEAT_PITCH + k] = f2bf(W[i]);
        }
        for (int n = tid; n < npad; n += 256) bias[n] = (n < dout) ? bg[L][n] : 0.f;
        __syncthreads();

        // ---- GEMM: [112 x kpad] @ [kpad x npad] via v_wmma_f32_16x16x32_bf16 ----
        const int nt_cnt = npad >> 4;
        const int kt_cnt = kpad >> 5;
        const int kbA = hi * 8;     // A: K chunks {kbA..+8, kbA+16..+8}
    const int kbB = hi * 8;     // B: K chunks {kbB..+8, kbB+16..+8}
        for (int t = wave; t < 7 * nt_cnt; t += 8) {
            int mt = t / nt_cnt, nt = t - mt * nt_cnt;
            v8f c = {};
            const unsigned short* arow = feat + (mt * 16 + lo) * FEAT_PITCH;
            const unsigned short* brow = wT   + (nt * 16 + lo) * FEAT_PITCH;
            for (int kt = 0; kt < kt_cnt; ++kt) {
                ABu a, b;
                a.q[0] = *(const uint4*)(arow + kt * 32 + kbA);
                a.q[1] = *(const uint4*)(arow + kt * 32 + kbA + 16);
                b.q[0] = *(const uint4*)(brow + kt * 32 + kbB);
                b.q[1] = *(const uint4*)(brow + kt * 32 + kbB + 16);
                c = __builtin_amdgcn_wmma_f32_16x16x32_bf16(
                        false, a.v, false, b.v, (short)0, c, false, false);
            }
            const int N  = nt * 16 + lo;
            const float bn = bias[N];
#pragma unroll
            for (int r = 0; r < 8; ++r) {
                int M = mt * 16 + hi * 8 + r;
                if (M < NPG && N < dout)
                    acc[M * ACC_PITCH + N] = fmaxf(c[r] + bn, 0.f);  // bias + relu
            }
        }
        __syncthreads();

        // ---- write back next-layer input (bf16), re-zero padding ----
        if (L < 4) {
            for (int i = tid; i < 112 * FEAT_PITCH / 2; i += 256) ((unsigned int*)feat)[i] = 0u;
            __syncthreads();
            for (int i = tid; i < NPG * dout; i += 256) {
                int m = i / dout, f = i - m * dout;
                feat[m * FEAT_PITCH + f] = f2bf(acc[m * ACC_PITCH + f]);
            }
            __syncthreads();
        }
    }

    // ---- gate + block-local softmax + attention pooling (whole graph in this block) ----
    const float gb = gate_b[0];
    for (int m = tid; m < NPG; m += 256) {
        float s = gb;
#pragma unroll
        for (int f = 0; f < 37; ++f) s += acc[m * ACC_PITCH + f] * gate_w[f];
        gate[m] = s;
    }
    __syncthreads();
    if (tid == 0) {
        float mx = -INFINITY;
        for (int m = 0; m < NPG; ++m) mx = fmaxf(mx, gate[m]);
        red[0] = mx;
    }
    __syncthreads();
    const float mx = red[0];
    for (int m = tid; m < NPG; m += 256) gate[m] = expf(gate[m] - mx);
    __syncthreads();
    if (tid == 0) {
        float z = 0.f;
        for (int m = 0; m < NPG; ++m) z += gate[m];
        red[1] = z;
    }
    __syncthreads();
    const float invz = 1.0f / red[1];
    for (int f = tid; f < 37; f += 256) {
        float s = 0.f;
        for (int m = 0; m < NPG; ++m) s += acc[m * ACC_PITCH + f] * gate[m];
        out[g * 37 + f] = s * invz;
    }
}

extern "C" void kernel_launch(void* const* d_in, const int* in_sizes, int n_in,
                              void* d_out, int out_size, void* d_ws, size_t ws_size,
                              hipStream_t stream) {
    (void)in_sizes; (void)n_in; (void)d_ws; (void)ws_size; (void)out_size;
    const float* x   = (const float*)d_in[0];
    const int*   src = (const int*)d_in[1];
    const int*   dst = (const int*)d_in[2];
    // d_in[3] = graph_ids (implicit: node/100), unused
    const float* W0 = (const float*)d_in[4],  *b0 = (const float*)d_in[5];
    const float* W1 = (const float*)d_in[6],  *b1 = (const float*)d_in[7];
    const float* W2 = (const float*)d_in[8],  *b2 = (const float*)d_in[9];
    const float* W3 = (const float*)d_in[10], *b3 = (const float*)d_in[11];
    const float* W4 = (const float*)d_in[12], *b4 = (const float*)d_in[13];
    const float* gw = (const float*)d_in[14], *gbv = (const float*)d_in[15];
    float* out = (float*)d_out;

    hipFuncSetAttribute((const void*)DrugGraphConv_53798760350030_kernel,
                        hipFuncAttributeMaxDynamicSharedMemorySize, SMEM_BYTES);

    DrugGraphConv_53798760350030_kernel<<<NG, 256, SMEM_BYTES, stream>>>(
        x, src, dst, W0, b0, W1, b1, W2, b2, W3, b3, W4, b4, gw, gbv, out);
}
